// MultiRelGraphConv_12326556140210
// MI455X (gfx1250) — compile-verified, matrix operand order checked by reading, not run on hardware
//
#include <hip/hip_runtime.h>

#define DD 128
#define NNODES 50000
#define NEDGES 800000

typedef __attribute__((ext_vector_type(16))) _Float16 v16h;
typedef __attribute__((ext_vector_type(8)))  float    v8f;

// ---------------------------------------------------------------------------
// Scatter: for edge e, atomically add feats[gidx?gidx[e]:e] row into S[dst[e]];
// one wave32 per edge, each lane handles 4 floats (float4). Optionally counts.
// ---------------------------------------------------------------------------
__global__ void scatter_rows_kernel(const float* __restrict__ feats,
                                    const int*   __restrict__ gidx,
                                    const int*   __restrict__ dst,
                                    float*       __restrict__ S,
                                    float*       __restrict__ cnt,
                                    int nE) {
    const int lane = threadIdx.x & 31;
    const int wave = threadIdx.x >> 5;
    const int e = blockIdx.x * (blockDim.x >> 5) + wave;
    if (e >= nE) return;
    const int srow = gidx ? gidx[e] : e;
    const int drow = dst[e];
    const float4 v = ((const float4*)(feats + (size_t)srow * DD))[lane];
    float* out = S + (size_t)drow * DD + lane * 4;
    unsafeAtomicAdd(out + 0, v.x);
    unsafeAtomicAdd(out + 1, v.y);
    unsafeAtomicAdd(out + 2, v.z);
    unsafeAtomicAdd(out + 3, v.w);
    if (cnt != nullptr && lane == 0) unsafeAtomicAdd(cnt + drow, 1.0f);
}

// ---------------------------------------------------------------------------
// One-time weight pack: f32 row-major [128x128] -> f16 WMMA B-fragment order.
// Fragment (ks, nt): 32 lanes x 16 f16; lane<16 holds K {0..7,16..23},
// lane>=16 holds K {8..15,24..31} (ISA 16-bit A/B VGPR map); N = nt*16 + lane%16.
// ---------------------------------------------------------------------------
__global__ void pack_weights_kernel(const float* __restrict__ W,
                                    _Float16* __restrict__ outp) {
    const int idx = blockIdx.x * blockDim.x + threadIdx.x;   // 0..16383
    if (idx >= DD * DD) return;
    const int k = idx >> 7, n = idx & 127;
    const int ks = k >> 5, kr = k & 31, g = kr >> 3;
    const int lane = ((g & 1) << 4) + (n & 15);
    const int elem = (kr & 7) + ((g >> 1) << 3);
    outp[(ks * 8 + (n >> 4)) * 512 + lane * 16 + elem] = (_Float16)W[idx];
}

// ---------------------------------------------------------------------------
// A-tile fragment: 8+8 consecutive f32 (K +0..7 and +16..23 from `off`),
// scaled and converted to f16 in WMMA A layout order.
// ---------------------------------------------------------------------------
__device__ __forceinline__ v16h load_a_frag(const float* __restrict__ rowPtr,
                                            int off, float scale) {
    const float4* p4 = (const float4*)(rowPtr + off);
    const float4 x0 = p4[0], x1 = p4[1];     // K +0..7
    const float4 x2 = p4[4], x3 = p4[5];     // K +16..23
    v16h a;
    a[0]=(_Float16)(x0.x*scale);  a[1]=(_Float16)(x0.y*scale);
    a[2]=(_Float16)(x0.z*scale);  a[3]=(_Float16)(x0.w*scale);
    a[4]=(_Float16)(x1.x*scale);  a[5]=(_Float16)(x1.y*scale);
    a[6]=(_Float16)(x1.z*scale);  a[7]=(_Float16)(x1.w*scale);
    a[8]=(_Float16)(x2.x*scale);  a[9]=(_Float16)(x2.y*scale);
    a[10]=(_Float16)(x2.z*scale); a[11]=(_Float16)(x2.w*scale);
    a[12]=(_Float16)(x3.x*scale); a[13]=(_Float16)(x3.y*scale);
    a[14]=(_Float16)(x3.z*scale); a[15]=(_Float16)(x3.w*scale);
    return a;
}

__device__ __forceinline__ void epilogue_store(const v8f* __restrict__ acc,
                                               int rbase, int lane,
                                               float* __restrict__ Cout,
                                               const float* __restrict__ bn,
                                               const float* __restrict__ bl,
                                               const float* __restrict__ cnt,
                                               int doRelu, int nRows, float slope) {
    const int colLane = lane & 15;
    const int rOff    = (lane >> 4) << 3;      // C/D map: lanes>=16 -> M+8
    float ind[8]; int rows[8];
    #pragma unroll
    for (int r = 0; r < 8; ++r) {
        const int row = rbase + r + rOff;
        rows[r] = row;
        float c = 1.0f;
        if (cnt != nullptr) {
            const int rc = (row < nRows) ? row : 0;
            c = (cnt[rc] > 0.0f) ? 1.0f : 0.0f;
        }
        ind[r] = c;
    }
    #pragma unroll
    for (int nt = 0; nt < 8; ++nt) {
        const int col = nt * 16 + colLane;
        const float bnv = (bn != nullptr) ? bn[col] : 0.0f;
        const float blv = bl[col];
        #pragma unroll
        for (int r = 0; r < 8; ++r) {
            if (rows[r] < nRows) {
                float v = acc[nt][r] + ind[r] * bnv + blv;
                if (doRelu) v = (v >= 0.0f) ? v : v * slope;
                Cout[(size_t)rows[r] * DD + col] = v;
            }
        }
    }
}

// ---------------------------------------------------------------------------
// Fused multi-segment GEMM + (1/deg scaling) + bias + RReLU.
// C[N x 128] = sum_seg (scale_seg ? A_seg/deg : A_seg) @ B_seg + ind*bn + bl
// Block: 128 threads = 4 waves; each wave owns 32 rows (two 16-row M-tiles),
// so every B fragment (direct 32B/lane global load, L2-broadcast of the
// pre-packed f16 weights) feeds two back-to-back v_wmma_f32_16x16x32_f16.
// ---------------------------------------------------------------------------
__global__ __launch_bounds__(128)
void gemm_wmma_kernel(const float* __restrict__ A0, const float* __restrict__ A1,
                      const float* __restrict__ A2,
                      const _Float16* __restrict__ Wpk,
                      int nseg, int scaleMask,
                      float* __restrict__ Cout,
                      const float* __restrict__ bn, const float* __restrict__ bl,
                      const float* __restrict__ cnt, int doRelu, int nRows,
                      float slope) {
    const int lane    = threadIdx.x & 31;
    const int wave    = threadIdx.x >> 5;
    const int rowBase = blockIdx.x * 128 + wave * 32;
    const int m0 = rowBase + (lane & 15);
    const int m1 = m0 + 16;
    const int m0C = (m0 < nRows) ? m0 : 0;     // clamp loads; stores masked
    const int m1C = (m1 < nRows) ? m1 : 0;
    const int hiOff = (lane >> 4) << 3;        // A map: lanes>=16 -> K+8

    float rc0 = 1.0f, rc1 = 1.0f;
    if (cnt != nullptr) {
        rc0 = 1.0f / fmaxf(cnt[m0C], 1.0f);
        rc1 = 1.0f / fmaxf(cnt[m1C], 1.0f);
    }

    v8f acc0[8] = {};
    v8f acc1[8] = {};

    for (int seg = 0; seg < nseg; ++seg) {
        const float* A = (seg == 0) ? A0 : (seg == 1) ? A1 : A2;
        const float s0 = ((scaleMask >> seg) & 1) ? rc0 : 1.0f;
        const float s1 = ((scaleMask >> seg) & 1) ? rc1 : 1.0f;
        const float* r0 = A + (size_t)m0C * DD;
        const float* r1 = A + (size_t)m1C * DD;
        if (seg + 1 < nseg) {                  // warm L1 for next segment's rows
            const float* An = (seg == 0) ? A1 : A2;
            __builtin_prefetch(An + (size_t)m0C * DD, 0, 1);
            __builtin_prefetch(An + (size_t)m1C * DD, 0, 1);
        }
        const _Float16* wseg = Wpk + seg * 16384 + lane * 16;
        #pragma unroll
        for (int ks = 0; ks < 4; ++ks) {
            const v16h a0 = load_a_frag(r0, ks * 32 + hiOff, s0);
            const v16h a1 = load_a_frag(r1, ks * 32 + hiOff, s1);
            const _Float16* wks = wseg + ks * 4096;
            #pragma unroll
            for (int nt = 0; nt < 8; ++nt) {
                const v16h b = *(const v16h*)(wks + nt * 512);
                acc0[nt] = __builtin_amdgcn_wmma_f32_16x16x32_f16(
                    false, a0, false, b, (short)0, acc0[nt], false, false);
                acc1[nt] = __builtin_amdgcn_wmma_f32_16x16x32_f16(
                    false, a1, false, b, (short)0, acc1[nt], false, false);
            }
        }
    }

    epilogue_store(acc0, rowBase,      lane, Cout, bn, bl, cnt, doRelu, nRows, slope);
    epilogue_store(acc1, rowBase + 16, lane, Cout, bn, bl, cnt, doRelu, nRows, slope);
}

// ---------------------------------------------------------------------------
// Host orchestration (graph-capture safe: only kernels + hipMemsetAsync)
// ---------------------------------------------------------------------------
extern "C" void kernel_launch(void* const* d_in, const int* in_sizes, int n_in,
                              void* d_out, int out_size, void* d_ws, size_t ws_size,
                              hipStream_t stream) {
    const float* node = (const float*)d_in[0];
    const float* edge = (const float*)d_in[1];
    const int*   src  = (const int*)d_in[2];
    const int*   dst  = (const int*)d_in[3];
    const float* Wn0  = (const float*)d_in[4];
    const float* bn0  = (const float*)d_in[5];
    const float* Wl0  = (const float*)d_in[6];
    const float* bl0  = (const float*)d_in[7];
    const float* Wn1  = (const float*)d_in[8];
    const float* bn1  = (const float*)d_in[9];
    const float* Wl1  = (const float*)d_in[10];
    const float* bl1  = (const float*)d_in[11];
    const float* Wo   = (const float*)d_in[12];
    const float* bo   = (const float*)d_in[13];
    float* out = (float*)d_out;

    float*     cnt   = (float*)d_ws;                      // [N]
    float*     Se    = cnt + NNODES;                      // [N*D] raw segsum(e) (layer-invariant)
    float*     Sh    = Se + (size_t)NNODES * DD;          // [N*D] raw segsum(h[src]); reused as h2
    float*     h1    = Sh + (size_t)NNODES * DD;          // [N*D]
    _Float16*  Wpack = (_Float16*)(h1 + (size_t)NNODES * DD);  // 8 segs * 16384 f16

    const float slope = (1.0f / 8.0f + 1.0f / 3.0f) * 0.5f;

    hipMemsetAsync(cnt, 0, sizeof(float) * NNODES, stream);
    hipMemsetAsync(Se,  0, sizeof(float) * (size_t)NNODES * DD, stream);
    hipMemsetAsync(Sh,  0, sizeof(float) * (size_t)NNODES * DD, stream);

    // pack all weight segments to f16 fragment order (L2-resident, 256 KB)
    const dim3 pBlk(256), pGrd((DD * DD + 255) / 256);
    pack_weights_kernel<<<pGrd, pBlk, 0, stream>>>(Wn0,           Wpack + 0 * 16384);
    pack_weights_kernel<<<pGrd, pBlk, 0, stream>>>(Wn0 + DD * DD, Wpack + 1 * 16384);
    pack_weights_kernel<<<pGrd, pBlk, 0, stream>>>(Wl0,           Wpack + 2 * 16384);
    pack_weights_kernel<<<pGrd, pBlk, 0, stream>>>(Wn1,           Wpack + 3 * 16384);
    pack_weights_kernel<<<pGrd, pBlk, 0, stream>>>(Wn1 + DD * DD, Wpack + 4 * 16384);
    pack_weights_kernel<<<pGrd, pBlk, 0, stream>>>(Wl1,           Wpack + 5 * 16384);
    pack_weights_kernel<<<pGrd, pBlk, 0, stream>>>(Wo,            Wpack + 6 * 16384);
    pack_weights_kernel<<<pGrd, pBlk, 0, stream>>>(Wo + DD * DD,  Wpack + 7 * 16384);

    const dim3 sBlk(256);                                 // 8 waves = 8 edges/block
    const dim3 sGrd((NEDGES + 7) / 8);
    const dim3 gBlk(128), gGrd((NNODES + 127) / 128);

    // layer-invariant edge aggregation: S_e = segsum(e), degree cnt
    scatter_rows_kernel<<<sGrd, sBlk, 0, stream>>>(edge, nullptr, dst, Se, cnt, NEDGES);
    scatter_rows_kernel<<<sGrd, sBlk, 0, stream>>>(node, src,     dst, Sh, nullptr, NEDGES);

    // layer 0: h1 = rrelu(Sh/deg@Wn0[:D] + Se/deg@Wn0[D:] + node@Wl0 + 1[deg>0]*bn0 + bl0)
    gemm_wmma_kernel<<<gGrd, gBlk, 0, stream>>>(Sh, Se, node, Wpack + 0 * 16384, 3, 0x3,
                                                h1, bn0, bl0, cnt, 1, NNODES, slope);

    // layer 1 aggregation of h1
    hipMemsetAsync(Sh, 0, sizeof(float) * (size_t)NNODES * DD, stream);
    scatter_rows_kernel<<<sGrd, sBlk, 0, stream>>>(h1, src, dst, Sh, nullptr, NEDGES);

    // layer 1: h2 (in-place over Sh; each block reads only its own rows first)
    gemm_wmma_kernel<<<gGrd, gBlk, 0, stream>>>(Sh, Se, h1, Wpack + 3 * 16384, 3, 0x3,
                                                Sh, bn1, bl1, cnt, 1, NNODES, slope);

    // output: out = h1@Wo[:D] + h2@Wo[D:] + bo
    gemm_wmma_kernel<<<gGrd, gBlk, 0, stream>>>(h1, Sh, nullptr, Wpack + 6 * 16384, 2, 0x0,
                                                out, nullptr, bo, nullptr, 0, NNODES, slope);
}
